// FourierBlock_67637144978394
// MI455X (gfx1250) — matrix operand order repaired and checked by reading, not compiled, not run on hardware
//
#include <hip/hip_runtime.h>

// ---------------------------------------------------------------------------
// FourierBlock for MI455X (gfx1250): selected-mode DFT as three bf16 WMMA GEMMs
//   K1: X[128,64]   = Afwd[128,2048] x Qbh[2048,64]      per (b,h), ksplit=2
//   K2: Z[32,128]   = Xmix[32,128]  x Wmix[128,128]      per (h,m)
//   K3: Y[64,2048]  = O2[64,128]    x Ainv[128,2048]     per (b,h), lsplit=4
// Static operands pre-swizzled into WMMA fragment layouts (32B/lane loads).
// Pure-copy global->LDS stages use GLOBAL_LOAD_ASYNC_TO_LDS_B128 (ASYNCcnt);
// stages that need f32->bf16 conversion or element transposes route via VGPRs.
// ---------------------------------------------------------------------------

typedef float   v8f   __attribute__((ext_vector_type(8)));
typedef float   f4    __attribute__((ext_vector_type(4)));
typedef __bf16  v16bf __attribute__((ext_vector_type(16)));
typedef __bf16  v8bf  __attribute__((ext_vector_type(8)));

#define DEVINL static __device__ __forceinline__

DEVINL v8f wmma_bf16(v16bf a, v16bf b, v8f c) {
  // D = A(16x32 bf16) x B(32x16 bf16) + C(16x16 f32)
  return __builtin_amdgcn_wmma_f32_16x16x32_bf16(
      /*neg_a=*/false, a, /*neg_b=*/false, b,
      /*c_mod=*/(short)0, c, /*reuse_a=*/false, /*reuse_b=*/false);
}

// Async 16B global->LDS copy (no VGPR data; tracked with ASYNCcnt).
DEVINL void async_copy16(const void* gptr, void* lptr) {
  unsigned lds = (unsigned)(size_t)lptr;   // low 32 bits of generic LDS ptr = LDS offset
  asm volatile("global_load_async_to_lds_b128 %0, %1, off"
               :: "v"(lds), "v"(gptr) : "memory");
}
DEVINL void wait_async0() {
  asm volatile("s_wait_asynccnt 0x0" ::: "memory");
}

// A-fragment (16x32, 16-bit): element e of the per-lane v16bf maps to
// K = e + 16*hi   with hi=lane>=16 (same contiguous order as B).
DEVINL int a_kk(int e, int hi) {
  return e + 16 * hi;
}
// B-fragment (32x16, 16-bit): lane holds column n=lane&15, rows
// K = e + 16*hi  (contiguous K per lane -> contiguous 32B when col-major).

// problem constants
constexpr int Bc = 32, Lc = 2048, Hc = 8, Ec = 64, Mc = 64;

// workspace layout (bytes)
constexpr size_t AFWD_OFF  = 0;                       // [mt8][kt64][lane32][e16] bf16
constexpr size_t AFWD_SZ   = (size_t)8 * 64 * 32 * 16 * 2;          // 512 KB
constexpr size_t AINV_OFF  = AFWD_OFF + AFWD_SZ;      // [kt4][nt128][lane32][e16] bf16
constexpr size_t AINV_SZ   = (size_t)4 * 128 * 32 * 16 * 2;         // 512 KB
constexpr size_t WFRAG_OFF = AINV_OFF + AINV_SZ;      // [h*64+m][kt4][nt8][lane32][e16] bf16
constexpr size_t WFRAG_SZ  = (size_t)512 * 4 * 8 * 32 * 16 * 2;     // 16 MB
constexpr size_t XWS_OFF   = WFRAG_OFF + WFRAG_SZ;    // [chunk2][h][m][b*128+c] f32
constexpr size_t XWS_SZ    = (size_t)2 * Hc * Mc * Bc * 128 * 4;    // 16 MB
constexpr size_t OWS_OFF   = XWS_OFF + XWS_SZ;        // [b][h][o64][r128] f32 (transposed)
constexpr size_t OWS_SZ    = (size_t)Bc * Hc * 64 * 128 * 4;        // 8 MB

// ---------------------------------------------------------------------------
// K0a: build DFT bases directly in WMMA fragment layouts (bf16).
//   Afwd rows r<64: cos(2*pi*index[r]*l/L); r>=64: -sin(2*pi*index[r-64]*l/L)
//   Ainv rows r<64: w_r cos(2*pi*r*l/L), w_0=1/L, w=2/L; r>=64: -w' sin(...),
//   w'_0=0 (irfft ignores imag of DC), w'=2/L.
// ---------------------------------------------------------------------------
__global__ void k0_basis(const int* __restrict__ index,
                         __bf16* __restrict__ afwd, __bf16* __restrict__ ainv) {
  int slot = blockIdx.x * 256 + threadIdx.x;   // 32768 slots total
  int lane = slot & 31, lo = lane & 15, hi = lane >> 4;
  const float STEP = 6.28318530717958647692f / 2048.0f;
  if (slot < 16384) {                          // forward basis: [mt8][kt64][lane]
    int t = slot >> 5, kt = t & 63, mt = t >> 6;
    int r = mt * 16 + lo;
    int freq = index[r & 63];
    bool isSin = (r >= 64);
#pragma unroll
    for (int e = 0; e < 16; ++e) {
      int l = kt * 32 + a_kk(e, hi);
      int p = (freq * l) & (Lc - 1);           // exact integer phase (L = 2^11)
      float ang = (float)p * STEP;
      float v = isSin ? -__sinf(ang) : __cosf(ang);
      afwd[(size_t)slot * 16 + e] = (__bf16)v;
    }
  } else {                                     // inverse basis: [kt4][nt128][lane]
    int s2 = slot - 16384;
    int t = s2 >> 5, nt = t & 127, kt = t >> 7;
    int l = nt * 16 + lo;
#pragma unroll
    for (int e = 0; e < 16; ++e) {
      int r = kt * 32 + hi * 16 + e;           // B-fragment K index
      int m = r & 63;
      int p = (m * l) & (Lc - 1);
      float ang = (float)p * STEP;
      float v;
      if (r < 64) v = ((r == 0) ? 1.0f : 2.0f) * (1.0f / 2048.0f) * __cosf(ang);
      else        v = (m == 0) ? 0.0f : -(2.0f / 2048.0f) * __sinf(ang);
      ainv[(size_t)s2 * 16 + e] = (__bf16)v;
    }
  }
}

// ---------------------------------------------------------------------------
// K0b: pre-swizzle complex-mix weights into B-fragments (bf16).
//   Wmix[k][n] (per h,m):  [ Wr  Wi ]      k=i / i+64, n=o / o+64
//                          [-Wi  Wr ]
// ---------------------------------------------------------------------------
__global__ void k0_wfrag(const float* __restrict__ wr, const float* __restrict__ wi,
                         __bf16* __restrict__ wf) {
  int slot = blockIdx.x * 256 + threadIdx.x;   // 524288 slots
  int lane = slot & 31, lo = lane & 15, hi = lane >> 4;
  int t = slot >> 5, nt = t & 7, t2 = t >> 3, kt = t2 & 3, hm = t2 >> 2;
  int h = hm >> 6, m = hm & 63;
#pragma unroll
  for (int e = 0; e < 16; ++e) {
    int k = kt * 32 + hi * 16 + e;
    int n = nt * 16 + lo;
    int i = k & 63, o = n & 63;
    size_t base = (((size_t)h * 64 + i) * 64 + o) * 64 + m;   // [H][E][E][M]
    float v = (k < 64) ? ((n < 64) ? wr[base] :  wi[base])
                       : ((n < 64) ? -wi[base] : wr[base]);
    wf[(size_t)slot * 16 + e] = (__bf16)v;
  }
}

// ---------------------------------------------------------------------------
// K1: forward DFT GEMM. Grid 512 = (b,h)x2 k-chunks; 128 threads = 4 waves.
// Wave w owns m-tiles {2w,2w+1} x all 4 n-tiles (8 accum tiles, 64 VGPRs).
// Q staged per 32-row k-tile into double-buffered LDS as transposed bf16
// [i][l] (stride 40: bank-conflict-free, 16B aligned) -> 1 barrier per tile,
// staging of tile k+1 overlaps the WMMA stream of tile k.
// ---------------------------------------------------------------------------
__global__ __launch_bounds__(128) void k1_forward(const float* __restrict__ q,
                                                  const __bf16* __restrict__ afwd,
                                                  float* __restrict__ xws) {
  __shared__ __bf16 qs[2][64 * 40];
  int bx = blockIdx.x;
  int chunk = bx & 1, bh = bx >> 1, h = bh & 7, b = bh >> 3;
  int tid = threadIdx.x, lane = tid & 31, w = tid >> 5;
  int lo = lane & 15, hi = lane >> 4;
  int g = tid >> 5, lrow = tid & 31;           // staging role: i-block, l-row

  auto stage = [&](int ktg, int buf) {
    size_t l = (size_t)ktg * 32 + lrow;
    const float* qp = q + (((size_t)b * Lc + l) * Hc + h) * Ec + g * 16;
    if (ktg < 63) __builtin_prefetch(qp + 32 * Hc * Ec, 0, 0);   // next k-tile
#pragma unroll
    for (int u = 0; u < 4; ++u) {
      f4 v = ((const f4*)qp)[u];
#pragma unroll
      for (int c4 = 0; c4 < 4; ++c4)
        qs[buf][(g * 16 + u * 4 + c4) * 40 + lrow] = (__bf16)v[c4];
    }
  };

  v8f acc[2][4];
#pragma unroll
  for (int im = 0; im < 2; ++im)
#pragma unroll
    for (int nt = 0; nt < 4; ++nt) acc[im][nt] = (v8f)0.0f;

  stage(chunk * 32, 0);
  __syncthreads();

  for (int kt = 0; kt < 32; ++kt) {
    int ktg = chunk * 32 + kt;
    if (kt + 1 < 32) stage(ktg + 1, (kt + 1) & 1);     // overlaps WMMAs below
    const __bf16* qb = qs[kt & 1];
    v16bf bfr[4];
#pragma unroll
    for (int nt = 0; nt < 4; ++nt) {
      const v8bf* bp = (const v8bf*)(qb + (nt * 16 + lo) * 40 + hi * 16);
      v8bf b0 = bp[0], b1 = bp[1];
#pragma unroll
      for (int e = 0; e < 8; ++e) { bfr[nt][e] = b0[e]; bfr[nt][8 + e] = b1[e]; }
    }
#pragma unroll
    for (int im = 0; im < 2; ++im) {
      int mt = w * 2 + im;
      v16bf afr = *(const v16bf*)(afwd + (((size_t)mt * 64 + ktg) * 32 + lane) * 16);
#pragma unroll
      for (int nt = 0; nt < 4; ++nt)
        acc[im][nt] = wmma_bf16(afr, bfr[nt], acc[im][nt]);
    }
    __syncthreads();   // publishes buf[(kt+1)&1]; frees buf[kt&1] for next stage
  }
  // epilogue: X row r -> (mode m = r&63, real/imag), col i -> c = i + 64*isImag
#pragma unroll
  for (int im = 0; im < 2; ++im)
#pragma unroll
    for (int nt = 0; nt < 4; ++nt)
#pragma unroll
      for (int j = 0; j < 8; ++j) {
        int r = 16 * (w * 2 + im) + j + 8 * hi;
        int m = r & 63;
        int c = 16 * nt + lo + 64 * (r >> 6);
        xws[(((size_t)chunk * Hc + h) * Mc + m) * (Bc * 128) + b * 128 + c] =
            acc[im][nt][j];
      }
}

// ---------------------------------------------------------------------------
// K2: per-(h,m) complex mix, Z[32,128] = Xmix[32,128] x Wmix[128,128].
// Grid 512; 128 threads = 4 waves; wave w owns n-tiles {2w,2w+1} x m-tiles 0..1.
// Both k-chunk partial tiles are pure copies -> async global->LDS (ASYNCcnt);
// the chunk reduction folds into the fragment gather (add co-executes).
// Epilogue writes Ows TRANSPOSED [b][h][o][r] so K3's staging is a pure copy.
// ---------------------------------------------------------------------------
__global__ __launch_bounds__(128) void k2_mix(const float* __restrict__ xws,
                                              const __bf16* __restrict__ wfrag,
                                              float* __restrict__ ows) {
  __shared__ float xsA[32 * 132];
  __shared__ float xsB[32 * 132];
  int bx = blockIdx.x, h = bx >> 6, m = bx & 63;
  int tid = threadIdx.x, lane = tid & 31, w = tid >> 5;
  int lo = lane & 15, hi = lane >> 4;
  // async stage of both chunk partials (16B granules, 16B-aligned both sides)
  {
    int b = tid >> 2, cb = (tid & 3) * 32;
    const float* p0 = xws + (((size_t)0 * Hc + h) * Mc + m) * (Bc * 128) + b * 128 + cb;
    const float* p1 = p0 + (size_t)Hc * Mc * Bc * 128;
    float* l0 = &xsA[b * 132 + cb];
    float* l1 = &xsB[b * 132 + cb];
#pragma unroll
    for (int u = 0; u < 8; ++u) {
      async_copy16(p0 + u * 4, l0 + u * 4);
      async_copy16(p1 + u * 4, l1 + u * 4);
    }
  }
  wait_async0();
  __syncthreads();

  v8f acc[2][2];
#pragma unroll
  for (int im = 0; im < 2; ++im)
#pragma unroll
    for (int jn = 0; jn < 2; ++jn) acc[im][jn] = (v8f)0.0f;

#pragma unroll
  for (int kt = 0; kt < 4; ++kt) {
    v16bf afr[2];
#pragma unroll
    for (int im = 0; im < 2; ++im) {
      int bb = 16 * im + lo;
#pragma unroll
      for (int e = 0; e < 16; ++e) {
        int idx = bb * 132 + kt * 32 + a_kk(e, hi);
        afr[im][e] = (__bf16)(xsA[idx] + xsB[idx]);   // chunk reduction here
      }
    }
#pragma unroll
    for (int jn = 0; jn < 2; ++jn) {
      int nt = w * 2 + jn;
      v16bf bfr = *(const v16bf*)(
          wfrag + ((((size_t)(h * 64 + m) * 4 + kt) * 8 + nt) * 32 + lane) * 16);
#pragma unroll
      for (int im = 0; im < 2; ++im)
        acc[im][jn] = wmma_bf16(afr[im], bfr, acc[im][jn]);
    }
  }
  // epilogue: Z row=b, col c -> (r = m + 64*isImag, o = c&63); layout [b][h][o][r]
#pragma unroll
  for (int im = 0; im < 2; ++im)
#pragma unroll
    for (int jn = 0; jn < 2; ++jn)
#pragma unroll
      for (int j = 0; j < 8; ++j) {
        int bb = 16 * im + j + 8 * hi;
        int c = 16 * (w * 2 + jn) + lo;
        int r = m + 64 * (c >= 64);
        ows[(((size_t)bb * Hc + h) * 64 + (c & 63)) * 128 + r] = acc[im][jn][j];
      }
}

// ---------------------------------------------------------------------------
// K3: inverse DFT GEMM, Y[64,2048] = O2[64,128] x Ainv[128,2048].
// Grid 1024 = (b,h) x 4 l-splits; 128 threads = 4 waves; wave covers 8 n-tiles.
// O2 arrives pre-transposed [o][r] -> staging is a pure async row copy.
// All 16 A-fragments built once from LDS, reused across the n-stream.
// ---------------------------------------------------------------------------
__global__ __launch_bounds__(128) void k3_inverse(const float* __restrict__ ows,
                                                  const __bf16* __restrict__ ainv,
                                                  float* __restrict__ out) {
  __shared__ float os[64 * 132];               // [o][r], padded stride
  int bx = blockIdx.x, ls = bx & 3, bh = bx >> 2, h = bh & 7, b = bh >> 3;
  int tid = threadIdx.x, lane = tid & 31, w = tid >> 5;
  int lo = lane & 15, hi = lane >> 4;
  // async pure-copy stage: 64 rows x 512B, 2 threads per row
  {
    int o = tid >> 1, half = tid & 1;
    const float* p = ows + (((size_t)b * Hc + h) * 64 + o) * 128 + half * 64;
    float* lp = &os[o * 132 + half * 64];
#pragma unroll
    for (int u = 0; u < 16; ++u) async_copy16(p + u * 4, lp + u * 4);
  }
  wait_async0();
  __syncthreads();
  // all A-fragments once
  v16bf afr[4][4];
#pragma unroll
  for (int mt = 0; mt < 4; ++mt)
#pragma unroll
    for (int kt = 0; kt < 4; ++kt) {
      int o = 16 * mt + lo;
#pragma unroll
      for (int e = 0; e < 16; ++e)
        afr[mt][kt][e] = (__bf16)os[o * 132 + kt * 32 + a_kk(e, hi)];
    }
  // stream over 8 n-tiles (128 columns of L)
  for (int ntl = 0; ntl < 8; ++ntl) {
    int nt = ls * 32 + w * 8 + ntl;
    v8f acc[4];
#pragma unroll
    for (int mt = 0; mt < 4; ++mt) acc[mt] = (v8f)0.0f;
#pragma unroll
    for (int kt = 0; kt < 4; ++kt) {
      v16bf bfr = *(const v16bf*)(ainv + (((size_t)kt * 128 + nt) * 32 + lane) * 16);
#pragma unroll
      for (int mt = 0; mt < 4; ++mt)
        acc[mt] = wmma_bf16(afr[mt][kt], bfr, acc[mt]);
    }
#pragma unroll
    for (int mt = 0; mt < 4; ++mt)
#pragma unroll
      for (int j = 0; j < 8; ++j) {
        int o = 16 * mt + j + 8 * hi;
        int l = 16 * nt + lo;
        out[(((size_t)b * Hc + h) * Ec + o) * Lc + l] = acc[mt][j];
      }
  }
}

// ---------------------------------------------------------------------------
extern "C" void kernel_launch(void* const* d_in, const int* in_sizes, int n_in,
                              void* d_out, int out_size, void* d_ws, size_t ws_size,
                              hipStream_t stream) {
  const float* q     = (const float*)d_in[0];
  // d_in[1]=k, d_in[2]=v unused by FourierBlock
  const float* w_re  = (const float*)d_in[3];
  const float* w_im  = (const float*)d_in[4];
  const int*   index = (const int*)d_in[5];
  float* out = (float*)d_out;

  char* ws = (char*)d_ws;
  __bf16* afwd  = (__bf16*)(ws + AFWD_OFF);
  __bf16* ainv  = (__bf16*)(ws + AINV_OFF);
  __bf16* wfrag = (__bf16*)(ws + WFRAG_OFF);
  float*  xws   = (float*)(ws + XWS_OFF);
  float*  ows   = (float*)(ws + OWS_OFF);

  k0_basis  <<<128,  256, 0, stream>>>(index, afwd, ainv);
  k0_wfrag  <<<2048, 256, 0, stream>>>(w_re, w_im, wfrag);
  k1_forward<<<512,  128, 0, stream>>>(q, afwd, xws);
  k2_mix    <<<512,  128, 0, stream>>>(xws, wfrag, ows);
  k3_inverse<<<1024, 128, 0, stream>>>(ows, ainv, out);
}
